// MoE_49993419325635
// MI455X (gfx1250) — matrix operand order repaired and checked by reading, not compiled, not run on hardware
//
#include <hip/hip_runtime.h>
#include <hip/hip_bf16.h>

// Problem constants (from reference): B=1, S=4096, D=1024, H=2816, E=8, K=2
#define SEQ    4096
#define DMODEL 1024
#define HDIM   2816
#define NEXP   8

typedef __attribute__((ext_vector_type(16))) __bf16 v16bf;
typedef __attribute__((ext_vector_type(8)))  __bf16 v8bf;
typedef __attribute__((ext_vector_type(4)))  __bf16 v4bf;
typedef __attribute__((ext_vector_type(8)))  float  v8f;
typedef __attribute__((ext_vector_type(4)))  int    v4i;

// --- CDNA5 async global->LDS path (guarded; falls back to register staging) ---
#if defined(__has_builtin)
#if __has_builtin(__builtin_amdgcn_global_load_async_to_lds_b128)
#define HAVE_ASYNC_LDS 1
#endif
#if __has_builtin(__builtin_amdgcn_s_wait_asynccnt)
#define HAVE_WAIT_ASYNC 1
#endif
#endif
#ifndef HAVE_ASYNC_LDS
#define HAVE_ASYNC_LDS 0
#endif
#ifndef HAVE_WAIT_ASYNC
#define HAVE_WAIT_ASYNC 0
#endif

#if HAVE_ASYNC_LDS
// Builtin signature (confirmed by clang diagnostic): param0 = v4i addrspace(1)*.
typedef __attribute__((address_space(1))) v4i GLv4i;
typedef __attribute__((address_space(3))) v4i LDSv4i;
// Generic global address == AS1 address; generic LDS address low 32 bits == AS3 offset.
#define AS1P(p) ((GLv4i*)(unsigned long long)(p))
#define AS3P(p) ((LDSv4i*)(unsigned int)(unsigned long long)(p))

__device__ __forceinline__ void async_wait_le8() {
#if HAVE_WAIT_ASYNC
    __builtin_amdgcn_s_wait_asynccnt(8);
#else
    asm volatile("s_wait_asynccnt 0x8" ::: "memory");
#endif
}
__device__ __forceinline__ void async_wait_0() {
#if HAVE_WAIT_ASYNC
    __builtin_amdgcn_s_wait_asynccnt(0);
#else
    asm volatile("s_wait_asynccnt 0x0" ::: "memory");
#endif
}
#endif

// ---------------------------------------------------------------------------
// Gate: logits = x @ w_gate + b_gate ; top-2 ; softmax over the 2 kept values
// ---------------------------------------------------------------------------
__global__ void moe_gate_kernel(const float* __restrict__ x,
                                const float* __restrict__ wg,
                                const float* __restrict__ bg,
                                float* __restrict__ gate) {
    __shared__ float wgs[DMODEL * NEXP];   // 32 KB of the 320 KB/WGP LDS
    for (int i = threadIdx.x; i < DMODEL * NEXP; i += blockDim.x) wgs[i] = wg[i];
    __syncthreads();

    int s = blockIdx.x * blockDim.x + threadIdx.x;   // token index
    float logit[NEXP];
#pragma unroll
    for (int e = 0; e < NEXP; ++e) logit[e] = bg[e];

    const float4* x4 = (const float4*)(x + (size_t)s * DMODEL);
    for (int d4 = 0; d4 < DMODEL / 4; ++d4) {
        float4 xv = x4[d4];
        int d = d4 * 4;
#pragma unroll
        for (int e = 0; e < NEXP; ++e) {
            logit[e] += xv.x * wgs[(d + 0) * NEXP + e];
            logit[e] += xv.y * wgs[(d + 1) * NEXP + e];
            logit[e] += xv.z * wgs[(d + 2) * NEXP + e];
            logit[e] += xv.w * wgs[(d + 3) * NEXP + e];
        }
    }
    // top-2 (track values & indices, compare-only -> no scratch)
    float m1 = logit[0]; int i1 = 0;
#pragma unroll
    for (int e = 1; e < NEXP; ++e)
        if (logit[e] > m1) { m1 = logit[e]; i1 = e; }
    float m2 = -3.0e38f; int i2 = -1;
#pragma unroll
    for (int e = 0; e < NEXP; ++e)
        if (e != i1 && logit[e] > m2) { m2 = logit[e]; i2 = e; }
    float e2 = __expf(m2 - m1);
    float inv = 1.0f / (1.0f + e2);
    float g1 = inv, g2 = e2 * inv;
#pragma unroll
    for (int e = 0; e < NEXP; ++e) {
        float g = (e == i1) ? g1 : ((e == i2) ? g2 : 0.0f);
        gate[(size_t)s * NEXP + e] = g;
    }
}

// ---------------------------------------------------------------------------
// fp32 -> bf16 bulk convert (activation matrix x)
// ---------------------------------------------------------------------------
__global__ void cvt_bf16_kernel(const float* __restrict__ src,
                                __bf16* __restrict__ dst) {
    int i = (blockIdx.x * blockDim.x + threadIdx.x) * 4;
    float4 v = *(const float4*)(src + i);
    v4bf o = { (__bf16)v.x, (__bf16)v.y, (__bf16)v.z, (__bf16)v.w };
    *(v4bf*)(dst + i) = o;
}

// ---------------------------------------------------------------------------
// Transpose + convert: src fp32 [R][C] -> dst bf16 [C][R]  (B operand N-major)
// ---------------------------------------------------------------------------
__global__ void transpose_cvt_kernel(const float* __restrict__ src,
                                     __bf16* __restrict__ dst,
                                     int R, int C) {
    __shared__ float tile[32][33];
    int c0 = blockIdx.x * 32;
    int r0 = blockIdx.y * 32;
#pragma unroll
    for (int i = 0; i < 4; ++i)
        tile[threadIdx.y + 8 * i][threadIdx.x] =
            src[(size_t)(r0 + threadIdx.y + 8 * i) * C + c0 + threadIdx.x];
    __syncthreads();
#pragma unroll
    for (int i = 0; i < 4; ++i)
        dst[(size_t)(c0 + threadIdx.y + 8 * i) * R + r0 + threadIdx.x] =
            (__bf16)tile[threadIdx.x][threadIdx.y + 8 * i];
}

// ---------------------------------------------------------------------------
// Tiled bf16 WMMA GEMM:  C[M,N] = A[M,K] * Bt[N,K]^T  (+bias, +epilogue)
//   MODE 0: Out = bf16( silu(acc + bias[n]) )              (h for GEMM2)
//   MODE 1: Out = gate[m,e] * (acc + bias[n])   fp32 write (first expert)
//   MODE 2: Out += gate[m,e] * (acc + bias[n])  fp32 accum
// Block: 128x128x64 stage, 256 threads = 8 waves, wave tile 32x64 (2x4 WMMAs
// per 32-K substep, 16 WMMAs per stage). Double-buffered LDS; async
// global->LDS pipeline (branchless steady state, last stage peeled).
// ---------------------------------------------------------------------------
#define BM 128
#define BN 128
#define BK 64
#define LDT 72   // padded LDS row stride (bf16): 144B, 16B-aligned, bank-skewed

template <int MODE>
__launch_bounds__(256)
__global__ void moe_gemm_kernel(const __bf16* __restrict__ A,   // [M,K]
                                const __bf16* __restrict__ Bt,  // [N,K]
                                const float* __restrict__ bias, // [N]
                                void* __restrict__ Out,
                                const float* __restrict__ gate, // [S,NEXP]
                                int M, int N, int Kd, int expert) {
    __shared__ alignas(16) __bf16 As[2][BM * LDT];
    __shared__ alignas(16) __bf16 Bs[2][BN * LDT];

    const int m0 = blockIdx.y * BM;
    const int n0 = blockIdx.x * BN;
    const int t = threadIdx.x;
    const int lane = t & 31;
    const int w = t >> 5;          // wave 0..7
    const int wm = w >> 1;         // 0..3 -> 32-row slice
    const int wn = w & 1;          // 0..1 -> 64-col slice
    const int l15 = lane & 15;
    const int lh  = lane >> 4;     // lane half -> K sub-range select

    // staging map: thread -> (row 0..127, col 0/32) of the 128x64 bf16 tile
    const int sr = t >> 1;
    const int sc = (t & 1) * 32;
    const int soff = sr * LDT + sc;

    v8f acc[2][4];
#pragma unroll
    for (int i = 0; i < 2; ++i)
#pragma unroll
        for (int j = 0; j < 4; ++j) acc[i][j] = (v8f)0.0f;

    const __bf16* Ag = A  + (size_t)(m0 + sr) * Kd + sc;
    const __bf16* Bg = Bt + (size_t)(n0 + sr) * Kd + sc;
    const int nk = Kd / BK;
    int cur = 0;

    // ---- one 64-K compute stage from LDS buffer (Ab,Bb) ----
    auto compute_stage = [&](const __bf16* Ab, const __bf16* Bb) {
#pragma unroll
        for (int ks = 0; ks < BK; ks += 32) {
            // A 16x32: lanes 0-15 K {0..7,16..23}; lanes 16-31 K {8..15,24..31}
            v16bf af[2];
#pragma unroll
            for (int i = 0; i < 2; ++i) {
                const __bf16* p = Ab + (wm * 32 + i * 16 + l15) * LDT + ks + lh * 8;
                v8bf lo = *(const v8bf*)p;
                v8bf hi = *(const v8bf*)(p + 16);
                af[i] = __builtin_shufflevector(lo, hi, 0, 1, 2, 3, 4, 5, 6, 7,
                                                8, 9, 10, 11, 12, 13, 14, 15);
            }
            // B 32x16: lane = column n; lanes 0-15 K {0..7,16..23}; lanes 16-31 K {8..15,24..31}
            v16bf bfz[4];
#pragma unroll
            for (int j = 0; j < 4; ++j) {
                const __bf16* p = Bb + (wn * 64 + j * 16 + l15) * LDT + ks + lh * 8;
                v8bf lo = *(const v8bf*)p;
                v8bf hi = *(const v8bf*)(p + 16);
                bfz[j] = __builtin_shufflevector(lo, hi, 0, 1, 2, 3, 4, 5, 6, 7,
                                                 8, 9, 10, 11, 12, 13, 14, 15);
            }
#pragma unroll
            for (int i = 0; i < 2; ++i)
#pragma unroll
                for (int j = 0; j < 4; ++j)
                    acc[i][j] = __builtin_amdgcn_wmma_f32_16x16x32_bf16(
                        false, af[i], false, bfz[j], (short)0, acc[i][j], false, false);
        }
    };

#if HAVE_ASYNC_LDS
    // ---- async global->LDS double-buffered pipeline (8 async ops/stage) ----
#pragma unroll
    for (int c = 0; c < 4; ++c) {
        __builtin_amdgcn_global_load_async_to_lds_b128(
            AS1P(Ag + c * 8), AS3P(&As[0][soff + c * 8]), 0, 0);
        __builtin_amdgcn_global_load_async_to_lds_b128(
            AS1P(Bg + c * 8), AS3P(&Bs[0][soff + c * 8]), 0, 0);
    }
    // steady state: branchless — always issue next stage, partial wait, compute
    for (int kt = 0; kt < nk - 1; ++kt) {
        const __bf16* Agn = Ag + (kt + 1) * BK;
        const __bf16* Bgn = Bg + (kt + 1) * BK;
#pragma unroll
        for (int c = 0; c < 4; ++c) {
            __builtin_amdgcn_global_load_async_to_lds_b128(
                AS1P(Agn + c * 8), AS3P(&As[cur ^ 1][soff + c * 8]), 0, 0);
            __builtin_amdgcn_global_load_async_to_lds_b128(
                AS1P(Bgn + c * 8), AS3P(&Bs[cur ^ 1][soff + c * 8]), 0, 0);
        }
        async_wait_le8();   // in-order completion => stage kt's 8 ops landed
        __syncthreads();
        compute_stage(&As[cur][0], &Bs[cur][0]);
        __syncthreads();
        cur ^= 1;
    }
    // peeled final stage
    async_wait_0();
    __syncthreads();
    compute_stage(&As[cur][0], &Bs[cur][0]);
#else
    // ---- fallback: pipelined register staging, double-buffered LDS ----
    v8bf sa[4], sb[4];
#pragma unroll
    for (int c = 0; c < 4; ++c) {
        sa[c] = *(const v8bf*)(Ag + c * 8);
        sb[c] = *(const v8bf*)(Bg + c * 8);
    }
    for (int kt = 0; kt < nk - 1; ++kt) {
#pragma unroll
        for (int c = 0; c < 4; ++c) {
            *(v8bf*)(&As[cur][soff + c * 8]) = sa[c];
            *(v8bf*)(&Bs[cur][soff + c * 8]) = sb[c];
        }
        const __bf16* Agn = Ag + (kt + 1) * BK;
        const __bf16* Bgn = Bg + (kt + 1) * BK;
#pragma unroll
        for (int c = 0; c < 4; ++c) {
            sa[c] = *(const v8bf*)(Agn + c * 8);
            sb[c] = *(const v8bf*)(Bgn + c * 8);
        }
        __syncthreads();
        compute_stage(&As[cur][0], &Bs[cur][0]);
        __syncthreads();
        cur ^= 1;
    }
#pragma unroll
    for (int c = 0; c < 4; ++c) {
        *(v8bf*)(&As[cur][soff + c * 8]) = sa[c];
        *(v8bf*)(&Bs[cur][soff + c * 8]) = sb[c];
    }
    __syncthreads();
    compute_stage(&As[cur][0], &Bs[cur][0]);
#endif

    // ---- epilogue (branchless: MODE is a template param) ----
    // C/D layout: VGPR r -> M = r + 8*laneHalf; N = lane&15
    float bv[4];
#pragma unroll
    for (int j = 0; j < 4; ++j) bv[j] = bias[n0 + wn * 64 + j * 16 + l15];

#pragma unroll
    for (int i = 0; i < 2; ++i) {
        const int mbase = m0 + wm * 32 + i * 16 + 8 * lh;
#pragma unroll
        for (int r = 0; r < 8; ++r) {
            const int mm = mbase + r;
            if (MODE == 0) {
                __bf16* Oh = (__bf16*)Out;
#pragma unroll
                for (int j = 0; j < 4; ++j) {
                    float v = acc[i][j][r] + bv[j];
                    float sv = v / (1.0f + __expf(-v));   // silu
                    Oh[(size_t)mm * N + n0 + wn * 64 + j * 16 + l15] = (__bf16)sv;
                }
            } else {
                const float g = gate[(size_t)mm * NEXP + expert];
                float* Of = (float*)Out;
#pragma unroll
                for (int j = 0; j < 4; ++j) {
                    float v = acc[i][j][r] + bv[j];
                    float* op = Of + (size_t)mm * N + n0 + wn * 64 + j * 16 + l15;
                    float prev = (MODE == 2) ? *op : 0.0f;
                    *op = prev + g * v;
                }
            }
        }
    }
    (void)M;
}

// ---------------------------------------------------------------------------
// Host-side orchestration (graph-capture safe: only kernel launches on stream)
// ---------------------------------------------------------------------------
extern "C" void kernel_launch(void* const* d_in, const int* in_sizes, int n_in,
                              void* d_out, int out_size, void* d_ws, size_t ws_size,
                              hipStream_t stream) {
    const float* x      = (const float*)d_in[0];   // [S,D]
    const float* w_gate = (const float*)d_in[1];   // [D,E]
    const float* b_gate = (const float*)d_in[2];   // [E]
    const float* w1     = (const float*)d_in[3];   // [E,D,H]
    const float* b1     = (const float*)d_in[4];   // [E,H]
    const float* w2     = (const float*)d_in[5];   // [E,H,D]
    const float* b2     = (const float*)d_in[6];   // [E,D]
    float* out = (float*)d_out;                    // [S,D]

    // Workspace layout (all chunks 16B-aligned; ~43 MB total)
    char* ws = (char*)d_ws;
    size_t off = 0;
    float*  gate = (float*)(ws + off); off += (size_t)SEQ * NEXP * sizeof(float);
    __bf16* xb   = (__bf16*)(ws + off); off += (size_t)SEQ * DMODEL * sizeof(__bf16);
    __bf16* w1t  = (__bf16*)(ws + off); off += (size_t)HDIM * DMODEL * sizeof(__bf16);
    __bf16* w2t  = (__bf16*)(ws + off); off += (size_t)DMODEL * HDIM * sizeof(__bf16);
    __bf16* h    = (__bf16*)(ws + off); off += (size_t)SEQ * HDIM * sizeof(__bf16);
    (void)in_sizes; (void)n_in; (void)out_size; (void)ws_size;

    moe_gate_kernel<<<SEQ / 256, 256, 0, stream>>>(x, w_gate, b_gate, gate);
    cvt_bf16_kernel<<<(SEQ * DMODEL / 4) / 256, 256, 0, stream>>>(x, xb);

    for (int e = 0; e < NEXP; ++e) {
        transpose_cvt_kernel<<<dim3(HDIM / 32, DMODEL / 32), dim3(32, 8), 0, stream>>>(
            w1 + (size_t)e * DMODEL * HDIM, w1t, DMODEL, HDIM);
        transpose_cvt_kernel<<<dim3(DMODEL / 32, HDIM / 32), dim3(32, 8), 0, stream>>>(
            w2 + (size_t)e * HDIM * DMODEL, w2t, HDIM, DMODEL);

        // h = silu(x @ w1[e] + b1[e]) : M=S, N=H, K=D
        moe_gemm_kernel<0><<<dim3(HDIM / BN, SEQ / BM), 256, 0, stream>>>(
            xb, w1t, b1 + (size_t)e * HDIM, (void*)h, (const float*)nullptr,
            SEQ, HDIM, DMODEL, e);

        // out (+)= gate[:,e] * (h @ w2[e] + b2[e]) : M=S, N=D, K=H
        if (e == 0)
            moe_gemm_kernel<1><<<dim3(DMODEL / BN, SEQ / BM), 256, 0, stream>>>(
                h, w2t, b2 + (size_t)e * DMODEL, (void*)out, gate,
                SEQ, DMODEL, HDIM, e);
        else
            moe_gemm_kernel<2><<<dim3(DMODEL / BN, SEQ / BM), 256, 0, stream>>>(
                h, w2t, b2 + (size_t)e * DMODEL, (void*)out, gate,
                SEQ, DMODEL, HDIM, e);
    }
}
